// KDABlock_6468220748444
// MI455X (gfx1250) — matrix operand order repaired and checked by reading, not compiled, not run on hardware
//
#include <hip/hip_runtime.h>

// KDA chunkwise forward for MI455X (gfx1250, wave32, WMMA + TDM).
// Grid: B*H*VS blocks (128), 256 threads (8 wave32). Each block owns one
// (batch, head, V-slice) and scans the 32 chunks serially, keeping the
// 128x32 f32 state slice resident in LDS. GEMMs use
// v_wmma_f32_16x16x32_bf16; the v-chunk tile is staged LDS-side by the
// Tensor Data Mover one chunk ahead (tensor_load_to_lds / s_wait_tensorcnt).

typedef __attribute__((ext_vector_type(16))) __bf16 v16bf;
typedef __attribute__((ext_vector_type(8)))  float  v8f;
typedef __attribute__((ext_vector_type(4)))  unsigned int u32x4;
typedef __attribute__((ext_vector_type(8)))  unsigned int u32x8;

#define B_      2
#define H_      16
#define T_      2048
#define KD      128
#define VD      128
#define BT      64
#define BC      16
#define NCB     4          // BT/BC sub-blocks
#define NCHUNK  32         // T/BT
#define VS      4          // V-dimension split across blocks
#define VT      32         // V per block (VD/VS)
#define NTHREADS 256
#define NWAVES   8

// ---------------- WMMA fragment helpers (wave32; C/D per ISA, A/B in contiguous K order) ----------------

// A-matrix (16x32 bf16), K order chosen here, not ISA table order: lanes 0-15 / 16-31 both hold M=lane&15;
// K halves: lanes<16 -> K 0..15, lanes>=16 -> K 16..31; slot s -> K = kb + s.
__device__ __forceinline__ int a_klocal(int s, int lane) {
  int kb = (lane & 16) ? 16 : 0;
  return kb + s;
}

// bf16 WMMA: NEG/NEG_HI only support C-modifiers for bf16 (ISA 7.12), so
// A/B negation args must be 0; operand negation is folded into fragment build.
__device__ __forceinline__ v8f wmma_bf16(v16bf a, v16bf b, v8f c) {
  return __builtin_amdgcn_wmma_f32_16x16x32_bf16(false, a, false, b, (short)0, c,
                                                 false, false);
}

// f32 source A fragment, with compile-time sign (sign flip exact in bf16).
template <int SIGN>
__device__ __forceinline__ v16bf loadA_f32(const float* src, int stride,
                                           int rowBase, int kBase, int lane) {
  v16bf a;
  const float* rp = src + (size_t)(rowBase + (lane & 15)) * stride + kBase;
#pragma unroll
  for (int s = 0; s < 16; ++s)
    a[s] = (__bf16)((float)SIGN * rp[a_klocal(s, lane)]);
  return a;
}

__device__ __forceinline__ v16bf loadA_bf16(const __bf16* src, int stride,
                                            int rowBase, int kBase, int lane) {
  v16bf a;
  const __bf16* rp = src + (size_t)(rowBase + (lane & 15)) * stride + kBase;
#pragma unroll
  for (int s = 0; s < 16; ++s) a[s] = rp[a_klocal(s, lane)];
  return a;
}

__device__ __forceinline__ v16bf loadA_bf16_scaled(const __bf16* src, int stride,
                                                   int rowBase, int kBase,
                                                   const float* colScale, int lane) {
  v16bf a;
  const __bf16* rp = src + (size_t)(rowBase + (lane & 15)) * stride + kBase;
  const float* cs = colScale + kBase;
#pragma unroll
  for (int s = 0; s < 16; ++s) {
    int kl = a_klocal(s, lane);
    a[s] = (__bf16)((float)rp[kl] * cs[kl]);
  }
  return a;
}

// A-fragment gathered from a transposed (K-major) source: A[m][kk] = src[kk][col+m]
__device__ __forceinline__ v16bf loadAt_bf16(const __bf16* src, int stride,
                                             int colBase, int kBase, int lane) {
  v16bf a;
  int m = lane & 15;
#pragma unroll
  for (int s = 0; s < 16; ++s) {
    int kl = a_klocal(s, lane);
    a[s] = src[(size_t)(kBase + kl) * stride + colBase + m];
  }
  return a;
}

// B-matrix (32x16 bf16, KxN), same K order as A: lane n = lane&15; lanes<16 -> K 0..15,
// lanes>=16 -> K 16..31; slot s -> K = kb + s.
__device__ __forceinline__ v16bf loadB_bf16(const __bf16* src, int stride,
                                            int kBase, int nBase, int lane) {
  v16bf b;
  int n  = lane & 15;
  int kb = (lane & 16) ? 16 : 0;
#pragma unroll
  for (int s = 0; s < 16; ++s)
    b[s] = src[(size_t)(kBase + kb + s) * stride + nBase + n];
  return b;
}

// B-matrix from a row-major MxK bf16 source transposed (rows become N), with a
// per-K f32 scale: B[K][n] = src[nRow+n][K] * scale[K]
__device__ __forceinline__ v16bf loadBt_bf16_scaled(const __bf16* src, int stride,
                                                    int nRowBase, int kBase,
                                                    const float* colScale, int lane) {
  v16bf b;
  int n  = lane & 15;
  int kb = (lane & 16) ? 16 : 0;
  const __bf16* rp = src + (size_t)(nRowBase + n) * stride;
#pragma unroll
  for (int s = 0; s < 16; ++s) {
    int kk = kBase + kb + s;
    b[s] = (__bf16)((float)rp[kk] * colScale[kk]);
  }
  return b;
}

// C/D f32 16x16: VGPR r holds row M = r + 8*(lane>=16), N = lane&15.
__device__ __forceinline__ v8f loadC_f32(const float* src, int stride,
                                         int mBase, int nBase, int lane) {
  v8f c;
  int n  = lane & 15;
  int mo = (lane & 16) ? 8 : 0;
#pragma unroll
  for (int r = 0; r < 8; ++r)
    c[r] = src[(size_t)(mBase + mo + r) * stride + nBase + n];
  return c;
}

// ---------------- Tensor Data Mover: 64x32 f32 tile -> contiguous LDS --------
// D# per CDNA5 ISA ch.8: group0 = {flags, lds_addr, global_addr, type=2},
// group1 = {data_size=4B, tensor_dim0=VD, tensor_dim1=T, tile 32x64,
// dim0_stride=VD}. Issued by one wave; completion via s_wait_tensorcnt.
__device__ __forceinline__ void tdm_load_v_tile(const float* gptr, void* ldsptr) {
  unsigned long long ga = (unsigned long long)gptr;
  unsigned lds = (unsigned)(unsigned long long)ldsptr;  // low 32 bits = LDS offset
  u32x4 g0;
  g0[0] = 1u;                                           // count=1 (valid user D#)
  g0[1] = lds;                                          // lds_addr (bytes)
  g0[2] = (unsigned)(ga & 0xffffffffu);                 // global_addr[31:0]
  g0[3] = (unsigned)((ga >> 32) & 0x01ffffffu) | (2u << 30);  // addr[56:32]|type=2
  const unsigned td0 = VD, td1 = T_, tile0 = VT, tile1 = BT, str0 = VD;
  u32x8 g1;
  g1[0] = (2u << 16);                                   // data_size=4B, mask=0
  g1[1] = (td0 & 0xffffu) << 16;                        // tensor_dim0[15:0]
  g1[2] = ((td0 >> 16) & 0xffffu) | ((td1 & 0xffffu) << 16);
  g1[3] = ((td1 >> 16) & 0xffffu) | (tile0 << 16);      // tile_dim0
  g1[4] = tile1 & 0xffffu;                              // tile_dim1, tile_dim2=0
  g1[5] = str0;                                         // dim0_stride[31:0]
  g1[6] = 0u;                                           // dim0_stride[47:32], ds1 lo
  g1[7] = 0u;
  asm volatile("tensor_load_to_lds %0, %1" :: "s"(g0), "s"(g1) : "memory");
}

// ---------------- shared memory (~153 KB, fits 320 KB WGP LDS) ----------------

struct KDAShared {
  float  S[KD][VT];        // 16 KB  state slice (f32, persistent across chunks)
  __bf16 Sb[KD][VT];       //  8 KB  bf16 mirror of S (WMMA B operand)
  __bf16 qeg[BT][KD];      // 16 KB  q_n * scale * exp(gc)
  __bf16 keg[BT][KD];      // 16 KB  k_n * exp(gc)
  __bf16 kinv[BT][KD];     // 16 KB  k_n * exp(-gc)
  float  w[BT][KD];        // 32 KB  rhs_k -> w after solve
  float  u[BT][VT];        //  8 KB  rhs_v -> u after solve
  __bf16 vb[BT][VT];       //  4 KB  bf16 v_i (WMMA B operand)
  float  M[BT][BT];        // 16 KB  unit-lower system matrix (f32 for solve)
  __bf16 Ab[BT][BT];       //  8 KB  Aqk (bf16 A operand)
  float  vstage[BT][VT];   //  8 KB  TDM staging for v tile
  float  refp[NCB][KD];    //  2 KB  exp(+g_ref_i)
  float  refm[NCB][KD];    //  2 KB  exp(-g_ref_i)
  float  eglast[KD];
  float  qsc[BT];
  float  ksc[BT];
  float  betas[BT];
};

__global__ __launch_bounds__(NTHREADS)
void kda_chunkwise_kernel(const float* __restrict__ q, const float* __restrict__ k,
                          const float* __restrict__ v, const float* __restrict__ g,
                          const float* __restrict__ beta,
                          float* __restrict__ o, float* __restrict__ Sout) {
  __shared__ KDAShared sm;
  const int tid  = threadIdx.x;
  const int lane = tid & 31;
  const int wave = tid >> 5;
  const int vs = blockIdx.x % VS;
  const int bh = blockIdx.x / VS;

  const size_t rowBase = (size_t)bh * T_;   // row index base (rows of KD / VD)
  const float  qk_scale = 0.08838834764831845f;  // 128^-0.5

  // zero state (f32 + bf16 mirror)
  for (int i = tid; i < KD * VT; i += NTHREADS) {
    (&sm.S[0][0])[i]  = 0.0f;
    (&sm.Sb[0][0])[i] = (__bf16)0.0f;
  }
  // TDM: stage v tile of chunk 0
  if (wave == 0)
    tdm_load_v_tile(v + rowBase * VD + (size_t)vs * VT, &sm.vstage[0][0]);
  __syncthreads();

  for (int n = 0; n < NCHUNK; ++n) {
    const size_t chunkRow = rowBase + (size_t)n * BT;

    // ---- Phase 1: per-row L2 norms of q,k and beta ----
    if (tid < BT) {
      const float* qr = q + (chunkRow + tid) * KD;
      const float* kr = k + (chunkRow + tid) * KD;
      float sq = 0.0f, sk = 0.0f;
      for (int j = 0; j < KD; ++j) {
        float a0 = qr[j]; sq += a0 * a0;
        float a1 = kr[j]; sk += a1 * a1;
      }
      sm.qsc[tid]   = qk_scale / (sqrtf(sq) + 1e-6f);
      sm.ksc[tid]   = 1.0f / (sqrtf(sk) + 1e-6f);
      sm.betas[tid] = beta[chunkRow + tid];
    }
    __syncthreads();

    // ---- Phase 2: per-column g cumsum + decay-scaled operand tiles ----
    if (tid < KD) {
      const int kk = tid;
      float run = 0.0f;
      for (int r = 0; r < BT; ++r) {
        size_t idx = (chunkRow + r) * KD + kk;
        run += g[idx];
        float e  = __expf(run);     // exp(gc) <= 1 (g increments < 0)
        float ei = __expf(-run);    // exp(-gc) >= 1
        float qn = q[idx] * sm.qsc[r];
        float kn = k[idx] * sm.ksc[r];
        sm.qeg[r][kk]  = (__bf16)(qn * e);
        sm.keg[r][kk]  = (__bf16)(kn * e);
        sm.kinv[r][kk] = (__bf16)(kn * ei);
        if ((r & (BC - 1)) == 0) { sm.refp[r >> 4][kk] = e; sm.refm[r >> 4][kk] = ei; }
        if (r == BT - 1) sm.eglast[kk] = e;
      }
    } else if (n + 1 < NCHUNK) {
      // other half of the block prefetches next chunk (global_prefetch_b8)
      int t2 = tid - KD;  // 0..127 -> 64 rows x 2 halves
      size_t nr = (chunkRow + BT + (t2 >> 1)) * KD + (size_t)(t2 & 1) * 64;
      __builtin_prefetch(q + nr, 0, 0);
      __builtin_prefetch(k + nr, 0, 0);
      __builtin_prefetch(g + nr, 0, 0);
    }
    if (wave == 0) __builtin_amdgcn_s_wait_tensorcnt(0);  // v tile staged
    __syncthreads();

    // ---- Phase 2c: build RHS [w|u] from staged v, clear M/Aqk ----
    for (int i = tid; i < BT * KD; i += NTHREADS) {
      int r = i >> 7, c = i & (KD - 1);
      sm.w[r][c] = sm.betas[r] * (float)sm.keg[r][c];     // beta * k * exp(gc)
    }
    for (int i = tid; i < BT * VT; i += NTHREADS) {
      int r = i >> 5, c = i & (VT - 1);
      sm.u[r][c] = sm.betas[r] * sm.vstage[r][c];
    }
    for (int i = tid; i < BT * BT; i += NTHREADS) {
      (&sm.M[0][0])[i]  = 0.0f;
      (&sm.Ab[0][0])[i] = (__bf16)0.0f;
    }
    __syncthreads();

    // TDM: start staging next chunk's v tile (vstage reads all done above)
    if (wave == 0 && n + 1 < NCHUNK)
      tdm_load_v_tile(v + (chunkRow + BT) * VD + (size_t)vs * VT, &sm.vstage[0][0]);

    // ---- Phase 3: decay-weighted Gram sub-blocks via WMMA ----
    // pair p -> (i, j), j <= i ; row operands scaled by exp(-g_ref_i),
    // column operands scaled by exp(+g_ref_i); diagonal blocks masked.
    for (int p = wave; p < 10; p += NWAVES) {
      int i = (p >= 6) ? 3 : (p >= 3) ? 2 : (p >= 1) ? 1 : 0;
      int j = p - (i * (i + 1)) / 2;
      v8f a0 = {};
      v8f aq = {};
#pragma unroll
      for (int kt = 0; kt < 4; ++kt) {
        int kb = kt * 32;
        v16bf bf = loadBt_bf16_scaled(&sm.kinv[0][0], KD, j * BC, kb, sm.refp[i], lane);
        v16bf aK = loadA_bf16_scaled(&sm.keg[0][0], KD, i * BC, kb, sm.refm[i], lane);
        v16bf aQ = loadA_bf16_scaled(&sm.qeg[0][0], KD, i * BC, kb, sm.refm[i], lane);
        a0 = wmma_bf16(aK, bf, a0);
        aq = wmma_bf16(aQ, bf, aq);
      }
      int nn = lane & 15;
      int mo = (lane & 16) ? 8 : 0;
#pragma unroll
      for (int r = 0; r < 8; ++r) {
        int m = mo + r;
        int row = i * BC + m, col = j * BC + nn;
        float x0 = a0[r], xq = aq[r];
        if (i == j) {
          if (m <= nn) x0 = 0.0f;   // strictly lower for A0
          if (m <  nn) xq = 0.0f;   // inclusive lower for Aqk
        }
        sm.M[row][col]  = x0 * sm.betas[row] + ((i == j && m == nn) ? 1.0f : 0.0f);
        sm.Ab[row][col] = (__bf16)xq;
      }
    }
    __syncthreads();

    // ---- Phase 4: exact f32 unit-lower triangular solve, in place on [w|u] ----
    if (tid < KD + VT) {
      const bool isW = tid < KD;
      const int  c   = isW ? tid : tid - KD;
      for (int r = 1; r < BT; ++r) {
        float x = isW ? sm.w[r][c] : sm.u[r][c];
        for (int p2 = 0; p2 < r; ++p2) {
          float xp = isW ? sm.w[p2][c] : sm.u[p2][c];
          x -= sm.M[r][p2] * xp;
        }
        if (isW) sm.w[r][c] = x; else sm.u[r][c] = x;
      }
    }
    __syncthreads();

    // ---- Phase 5: v_i = u - w * S  (8 tiles, one per wave) -> vb (bf16) ----
    {
      int ti = wave >> 1, tj = wave & 1;
      v8f acc = loadC_f32(&sm.u[0][0], VT, ti * BC, tj * BC, lane);
#pragma unroll
      for (int kt = 0; kt < 4; ++kt) {
        int kb = kt * 32;
        v16bf af = loadA_f32<-1>(&sm.w[0][0], KD, ti * BC, kb, lane);  // -w
        v16bf bS = loadB_bf16(&sm.Sb[0][0], VT, kb, tj * BC, lane);
        acc = wmma_bf16(af, bS, acc);                    // u + (-w)*S
      }
      int nn = lane & 15, mo = (lane & 16) ? 8 : 0;
#pragma unroll
      for (int r = 0; r < 8; ++r)
        sm.vb[ti * BC + mo + r][tj * BC + nn] = (__bf16)acc[r];
    }
    __syncthreads();

    // ---- Phase 6: o = qeg * S + Aqk * v_i  -> global ----
    {
      int ti = wave >> 1, tj = wave & 1;
      v8f acc = {};
#pragma unroll
      for (int kt = 0; kt < 4; ++kt) {
        int kb = kt * 32;
        v16bf aq = loadA_bf16(&sm.qeg[0][0], KD, ti * BC, kb, lane);
        v16bf bS = loadB_bf16(&sm.Sb[0][0], VT, kb, tj * BC, lane);
        acc = wmma_bf16(aq, bS, acc);
      }
#pragma unroll
      for (int kt = 0; kt < 2; ++kt) {
        int kb = kt * 32;
        v16bf aA = loadA_bf16(&sm.Ab[0][0], BT, ti * BC, kb, lane);
        v16bf bV = loadB_bf16(&sm.vb[0][0], VT, kb, tj * BC, lane);
        acc = wmma_bf16(aA, bV, acc);
      }
      int nn = lane & 15, mo = (lane & 16) ? 8 : 0;
#pragma unroll
      for (int r = 0; r < 8; ++r) {
        size_t oidx = (chunkRow + ti * BC + mo + r) * VD + (size_t)vs * VT + tj * BC + nn;
        o[oidx] = acc[r];
      }
    }
    __syncthreads();

    // ---- Phase 7: S = diag(eglast) * (S + kinv^T * v_i)  (16 tiles) ----
    for (int t = wave; t < 16; t += NWAVES) {
      int ki = t >> 1, tj = t & 1;
      v8f acc = loadC_f32(&sm.S[0][0], VT, ki * BC, tj * BC, lane);
#pragma unroll
      for (int kt = 0; kt < 2; ++kt) {
        int kb = kt * 32;
        v16bf af = loadAt_bf16(&sm.kinv[0][0], KD, ki * BC, kb, lane);
        v16bf bV = loadB_bf16(&sm.vb[0][0], VT, kb, tj * BC, lane);
        acc = wmma_bf16(af, bV, acc);
      }
      int nn = lane & 15, mo = (lane & 16) ? 8 : 0;
#pragma unroll
      for (int r = 0; r < 8; ++r) {
        int krow = ki * BC + mo + r;
        float sv = acc[r] * sm.eglast[krow];
        sm.S[krow][tj * BC + nn]  = sv;
        sm.Sb[krow][tj * BC + nn] = (__bf16)sv;
      }
    }
    __syncthreads();
  }

  // ---- final state slice -> Sout (B,H,K,V) ----
  for (int i = tid; i < KD * VT; i += NTHREADS) {
    int kk = i >> 5, vc = i & (VT - 1);
    Sout[((size_t)bh * KD + kk) * VD + (size_t)vs * VT + vc] = sm.S[kk][vc];
  }
}

extern "C" void kernel_launch(void* const* d_in, const int* in_sizes, int n_in,
                              void* d_out, int out_size, void* d_ws, size_t ws_size,
                              hipStream_t stream) {
  (void)in_sizes; (void)n_in; (void)d_ws; (void)ws_size; (void)out_size;
  const float* q    = (const float*)d_in[0];
  const float* k    = (const float*)d_in[1];
  const float* v    = (const float*)d_in[2];
  const float* g    = (const float*)d_in[3];
  const float* beta = (const float*)d_in[4];
  float* o    = (float*)d_out;
  float* Sout = o + (size_t)B_ * H_ * T_ * VD;

  dim3 grid(B_ * H_ * VS);
  dim3 block(NTHREADS);
  hipLaunchKernelGGL(kda_chunkwise_kernel, grid, block, 0, stream,
                     q, k, v, g, beta, o, Sout);
}
